// multi_head_attention_30588757082715
// MI455X (gfx1250) — matrix operand order repaired and checked by reading, not compiled, not run on hardware
//
#include <hip/hip_runtime.h>
#include <hip/hip_bf16.h>

// ---- problem constants -----------------------------------------------------
constexpr int B = 4, S = 2048, D = 1024, H = 8, E = 128;

// ---- vector types for WMMA -------------------------------------------------
typedef __bf16 bf16_t;
typedef __attribute__((ext_vector_type(16))) __bf16 v16bf;
typedef __attribute__((ext_vector_type(8)))  __bf16 v8bf;
typedef __attribute__((ext_vector_type(8)))  float  v8f;

union V16 { v16bf v; v8bf h[2]; __bf16 e[16]; };
union V8F { v8f v; float e[8]; };

__device__ __forceinline__ v8f wmma_bf16(v16bf a, v16bf b, v8f c) {
    // D = A(16x32 bf16) * B(32x16 bf16) + C(16x16 f32)
    return __builtin_amdgcn_wmma_f32_16x16x32_bf16(
        false, a, false, b, (short)0, c, false, false);
}

// ============================================================================
// Kernel 0: one-time weight pre-transpose into bf16 "B-fragment native"
// layout Wt[n][k] (k contiguous per output column), so GEMM B-fragments are
// contiguous 32B global loads out of L2.
//   z in [0,24): head matrices  src = W{q,k,v}[h] (DxE) -> dst rows h*E..+127
//   z == 24:     Wo (DxD)       dst[n][k] = Wo[k][n]
// ============================================================================
__global__ __launch_bounds__(256)
void wtrans_kernel(const float* __restrict__ Wq, const float* __restrict__ Wk,
                   const float* __restrict__ Wv, const float* __restrict__ Wo,
                   bf16_t* __restrict__ wtq, bf16_t* __restrict__ wtk,
                   bf16_t* __restrict__ wtv, bf16_t* __restrict__ wto)
{
    const int z = blockIdx.z;
    const float* src; bf16_t* dst; int Nn;
    if (z < 24) {
        const int m = z >> 3, h = z & 7;
        src = ((m == 0) ? Wq : (m == 1) ? Wk : Wv) + (size_t)h * D * E;
        dst = ((m == 0) ? wtq : (m == 1) ? wtk : wtv) + (size_t)h * E * D;
        Nn = E;                         // src is D x E
    } else {
        src = Wo; dst = wto; Nn = D;    // src is D x D
    }
    const int bx = blockIdx.x, by = blockIdx.y;
    if (bx * 32 >= Nn) return;

    __shared__ float tile[32][33];
    const int tx = threadIdx.x, ty = threadIdx.y;   // 32 x 8
    #pragma unroll
    for (int j = 0; j < 4; ++j)
        tile[ty + 8 * j][tx] = src[(size_t)(by * 32 + ty + 8 * j) * Nn + bx * 32 + tx];
    __syncthreads();
    #pragma unroll
    for (int j = 0; j < 4; ++j)
        dst[(size_t)(bx * 32 + ty + 8 * j) * D + by * 32 + tx] =
            (bf16_t)tile[tx][ty + 8 * j];
}

// ============================================================================
// Kernel 1: fused Q/K/V projection, no LDS.
//   X[B*S, D] (f32) x Wt[n][k] (bf16) -> Yq/Yk bf16 [B,H,S,E],
//   Yv TRANSPOSED bf16 [B,H,E,S] so attention V B-frags are contiguous.
// Block: 128 thr (4 waves); wave tile 32(M) x 64(N); BK=32.
// ============================================================================
__global__ __launch_bounds__(128)
void proj_kernel(const float* __restrict__ q, const float* __restrict__ k,
                 const float* __restrict__ v,
                 const bf16_t* __restrict__ wtq, const bf16_t* __restrict__ wtk,
                 const bf16_t* __restrict__ wtv,
                 bf16_t* __restrict__ qh, bf16_t* __restrict__ kh,
                 bf16_t* __restrict__ vt)
{
    const int which = blockIdx.z;
    const float*  X  = (which == 0) ? q : (which == 1) ? k : v;
    const bf16_t* Wt = (which == 0) ? wtq : (which == 1) ? wtk : wtv;

    const int n0 = blockIdx.x * 64;            // cols within H*E
    const int m0 = blockIdx.y * 128;           // rows within B*S
    const int tid = threadIdx.x, lane = tid & 31, wid = tid >> 5;
    const int ln = lane & 15, grp = lane >> 4;
    const int mrow = m0 + wid * 32;            // this wave's 32 rows

    V8F acc[2][4];
    #pragma unroll
    for (int st = 0; st < 2; ++st)
        #pragma unroll
        for (int bt = 0; bt < 4; ++bt)
            #pragma unroll
            for (int r = 0; r < 8; ++r) acc[st][bt].e[r] = 0.0f;

    #pragma unroll 2
    for (int k0 = 0; k0 < D; k0 += 32) {
        // A fragments (f32 -> bf16):
        // lane<16: k[0..15]; lane>=16: k[16..31]
        V16 af[2];
        #pragma unroll
        for (int st = 0; st < 2; ++st) {
            const float* ap = X + (size_t)(mrow + st * 16 + ln) * D + k0 + grp * 16;
            #pragma unroll
            for (int j = 0; j < 8; ++j) {
                af[st].e[j]     = (bf16_t)ap[j];
                af[st].e[8 + j] = (bf16_t)ap[8 + j];
            }
        }
        // B fragments straight from transposed weights (contiguous 32B)
        #pragma unroll
        for (int bt = 0; bt < 4; ++bt) {
            const v16bf bfr = *(const v16bf*)
                (Wt + (size_t)(n0 + bt * 16 + ln) * D + k0 + grp * 16);
            acc[0][bt].v = wmma_bf16(af[0].v, bfr, acc[0][bt].v);
            acc[1][bt].v = wmma_bf16(af[1].v, bfr, acc[1][bt].v);
        }
    }

    // ---- store ----
    const int h  = n0 >> 7, e0 = n0 & 127;
    const int bb = m0 >> 11;                              // batch (S=2048)
    #pragma unroll
    for (int st = 0; st < 2; ++st) {
        const int sbase = (m0 & 2047) + wid * 32 + st * 16 + grp * 8;
        if (which < 2) {
            bf16_t* Y = (which == 0) ? qh : kh;
            const size_t base = (size_t)(bb * H + h) * S * E;
            #pragma unroll
            for (int bt = 0; bt < 4; ++bt) {
                const int e = e0 + bt * 16 + ln;
                #pragma unroll
                for (int r = 0; r < 8; ++r)
                    Y[base + (size_t)(sbase + r) * E + e] = (bf16_t)acc[st][bt].e[r];
            }
        } else {
            // V transposed: [B,H,E,S]; 8 rows contiguous in s -> b128 store
            #pragma unroll
            for (int bt = 0; bt < 4; ++bt) {
                const int e = e0 + bt * 16 + ln;
                V16 pk;
                #pragma unroll
                for (int r = 0; r < 8; ++r) pk.e[r] = (bf16_t)acc[st][bt].e[r];
                const size_t addr = ((size_t)(bb * H + h) * E + e) * S + sbase;
                *(v8bf*)(vt + addr) = pk.h[0];
            }
        }
    }
}

// ============================================================================
// Kernel 2: flash attention. One wave owns 16 query rows, streams 32 keys/iter.
// Q/K/V frags load directly from global bf16; score->A-frag transpose goes
// through a 1KB per-wave LDS region. Output z bf16 [B, S, H*E].
// ============================================================================
__global__ __launch_bounds__(128)
void attn_kernel(const bf16_t* __restrict__ qh, const bf16_t* __restrict__ kh,
                 const bf16_t* __restrict__ vt, bf16_t* __restrict__ z)
{
    const int bh  = blockIdx.y;                  // b*H + h
    const int tid = threadIdx.x, lane = tid & 31, wid = tid >> 5;
    const int ln = lane & 15, grp = lane >> 4;

    __shared__ bf16_t Ps[4][16 * 32];            // per-wave P tile, row-major [row][key]

    const bf16_t* Q = qh + (size_t)bh * S * E;
    const bf16_t* K = kh + (size_t)bh * S * E;
    const bf16_t* V = vt + (size_t)bh * E * S;   // [E][S]

    const int q0 = blockIdx.x * 64 + wid * 16;

    // preload Q A-fragments (K-dim E=128 -> 4 chunks of 32)
    V16 qf[4];
    {
        const bf16_t* qp = Q + (size_t)(q0 + ln) * E + grp * 16;
        #pragma unroll
        for (int c = 0; c < 4; ++c) {
            qf[c].h[0] = *(const v8bf*)(qp + c * 32);
            qf[c].h[1] = *(const v8bf*)(qp + c * 32 + 8);
        }
    }

    // softmax in exp2 domain: fold (1/sqrt(E)) * log2(e) into scores
    const float cexp = 0.08838834764831845f * 1.4426950408889634f;

    V8F   o[8];
    float mrow[8], lrow[8];
    #pragma unroll
    for (int t = 0; t < 8; ++t) {
        mrow[t] = -3.0e38f; lrow[t] = 0.0f;
        #pragma unroll
        for (int r = 0; r < 8; ++r) o[t].e[r] = 0.0f;
    }

    #pragma unroll 1
    for (int kb = 0; kb < S; kb += 32) {
        if (kb + 32 < S) {
            __builtin_prefetch(K + (size_t)(kb + 32 + ln) * E, 0, 0);
            __builtin_prefetch(V + (size_t)(ln * 8) * S + kb + 32, 0, 0);
        }

        // scores = Q . K^T  (2 key tiles of 16)
        V8F sc[2];
        #pragma unroll
        for (int t = 0; t < 2; ++t)
            #pragma unroll
            for (int r = 0; r < 8; ++r) sc[t].e[r] = 0.0f;

        #pragma unroll
        for (int c = 0; c < 4; ++c) {
            #pragma unroll
            for (int t = 0; t < 2; ++t) {
                const v16bf kp = *(const v16bf*)
                    (K + (size_t)(kb + t * 16 + ln) * E + c * 32 + grp * 16);
                sc[t].v = wmma_bf16(qf[c].v, kp, sc[t].v);
            }
        }

        // online softmax row update (rows live in 16-lane groups)
        float nm[8], al[8];
        #pragma unroll
        for (int r = 0; r < 8; ++r) {
            float s0 = sc[0].e[r] * cexp, s1 = sc[1].e[r] * cexp;
            sc[0].e[r] = s0; sc[1].e[r] = s1;
            float mx = fmaxf(s0, s1);
            mx = fmaxf(mx, __shfl_xor(mx, 1));
            mx = fmaxf(mx, __shfl_xor(mx, 2));
            mx = fmaxf(mx, __shfl_xor(mx, 4));
            mx = fmaxf(mx, __shfl_xor(mx, 8));
            nm[r] = fmaxf(mrow[r], mx);
            al[r] = exp2f(mrow[r] - nm[r]);
            mrow[r] = nm[r];
        }
        #pragma unroll
        for (int r = 0; r < 8; ++r) {
            float p0 = exp2f(sc[0].e[r] - nm[r]);
            float p1 = exp2f(sc[1].e[r] - nm[r]);
            sc[0].e[r] = p0; sc[1].e[r] = p1;
            float sm = p0 + p1;
            sm += __shfl_xor(sm, 1);
            sm += __shfl_xor(sm, 2);
            sm += __shfl_xor(sm, 4);
            sm += __shfl_xor(sm, 8);
            lrow[r] = lrow[r] * al[r] + sm;
        }
        #pragma unroll
        for (int et = 0; et < 8; ++et)
            #pragma unroll
            for (int r = 0; r < 8; ++r) o[et].e[r] *= al[r];

        // P (C layout) -> A-fragment layout via LDS
        bf16_t* ps = Ps[wid];
        const int c0 = ln;                       // contiguous: [0..31]
        const int c1 = c0 + 16;
        #pragma unroll
        for (int r = 0; r < 8; ++r) {
            const int row = grp * 8 + r;
            ps[row * 32 + c0] = (bf16_t)sc[0].e[r];
            ps[row * 32 + c1] = (bf16_t)sc[1].e[r];
        }
        asm volatile("s_wait_dscnt 0" ::: "memory");
        V16 pf;
        pf.v = *(const v16bf*)&ps[ln * 32 + grp * 16];

        // O += P . V  (V B-frags contiguous thanks to [E][S] layout)
        #pragma unroll
        for (int et = 0; et < 8; ++et) {
            const v16bf vp = *(const v16bf*)
                (V + (size_t)(et * 16 + ln) * S + kb + grp * 16);
            o[et].v = wmma_bf16(pf.v, vp, o[et].v);
        }
    }

    // normalize and store z[b, s, h*E + e] (bf16)
    const int bb = bh >> 3, hh = bh & 7;
    #pragma unroll
    for (int r = 0; r < 8; ++r) {
        const float rinv = 1.0f / lrow[r];
        const int s = q0 + grp * 8 + r;
        const size_t rowbase = ((size_t)bb * S + s) * (H * E) + hh * E;
        #pragma unroll
        for (int et = 0; et < 8; ++et)
            z[rowbase + et * 16 + ln] = (bf16_t)(o[et].e[r] * rinv);
    }
}

// ============================================================================
// Kernel 3: output projection  out = z(bf16) @ Wo + bo, out f32. No LDS;
// A-frags and B-frags (pre-transposed wto) direct from global.
// Block: 128 thr (4 waves); wave tile 32(M) x 64(N).
// ============================================================================
__global__ __launch_bounds__(128)
void outproj_kernel(const bf16_t* __restrict__ zin, const bf16_t* __restrict__ wto,
                    const float* __restrict__ bo, float* __restrict__ out)
{
    const int n0 = blockIdx.x * 64, m0 = blockIdx.y * 128;
    const int tid = threadIdx.x, lane = tid & 31, wid = tid >> 5;
    const int ln = lane & 15, grp = lane >> 4;
    const int mrow = m0 + wid * 32;

    V8F acc[2][4];
    #pragma unroll
    for (int st = 0; st < 2; ++st)
        #pragma unroll
        for (int bt = 0; bt < 4; ++bt)
            #pragma unroll
            for (int r = 0; r < 8; ++r) acc[st][bt].e[r] = 0.0f;

    #pragma unroll 2
    for (int k0 = 0; k0 < H * E; k0 += 32) {
        V16 af[2];
        #pragma unroll
        for (int st = 0; st < 2; ++st) {
            const bf16_t* ap = zin + (size_t)(mrow + st * 16 + ln) * (H * E)
                               + k0 + grp * 16;
            af[st].h[0] = *(const v8bf*)ap;
            af[st].h[1] = *(const v8bf*)(ap + 8);
        }
        #pragma unroll
        for (int bt = 0; bt < 4; ++bt) {
            const v16bf bfr = *(const v16bf*)
                (wto + (size_t)(n0 + bt * 16 + ln) * (H * E) + k0 + grp * 16);
            acc[0][bt].v = wmma_bf16(af[0].v, bfr, acc[0][bt].v);
            acc[1][bt].v = wmma_bf16(af[1].v, bfr, acc[1][bt].v);
        }
    }

    #pragma unroll
    for (int st = 0; st < 2; ++st)
        #pragma unroll
        for (int bt = 0; bt < 4; ++bt) {
            const int n = n0 + bt * 16 + ln;
            const float bias = bo[n];
            #pragma unroll
            for (int r = 0; r < 8; ++r) {
                const int row = mrow + st * 16 + grp * 8 + r;
                out[(size_t)row * D + n] = acc[st][bt].e[r] + bias;
            }
        }
}

// ============================================================================
extern "C" void kernel_launch(void* const* d_in, const int* in_sizes, int n_in,
                              void* d_out, int out_size, void* d_ws, size_t ws_size,
                              hipStream_t stream)
{
    const float* q  = (const float*)d_in[0];
    const float* k  = (const float*)d_in[1];
    const float* v  = (const float*)d_in[2];
    const float* Wq = (const float*)d_in[3];
    const float* Wk = (const float*)d_in[4];
    const float* Wv = (const float*)d_in[5];
    const float* Wo = (const float*)d_in[6];
    const float* bo = (const float*)d_in[7];
    float* out = (float*)d_out;

    const size_t nBHSE = (size_t)B * H * S * E;     // 8M elements
    const size_t nW    = (size_t)D * H * E;         // 1M elements
    bf16_t* qh  = (bf16_t*)d_ws;                    // [B,H,S,E]   16MB
    bf16_t* kh  = qh  + nBHSE;                      // [B,H,S,E]   16MB
    bf16_t* vt  = kh  + nBHSE;                      // [B,H,E,S]   16MB (transposed)
    bf16_t* z   = vt  + nBHSE;                      // [B,S,H*E]   16MB
    bf16_t* wtq = z   + nBHSE;                      // [H*E, D]     2MB
    bf16_t* wtk = wtq + nW;
    bf16_t* wtv = wtk + nW;
    bf16_t* wto = wtv + nW;

    // 0) one-time weight transpose into B-fragment-native bf16 layout
    wtrans_kernel<<<dim3(32, 32, 25), dim3(32, 8), 0, stream>>>(
        Wq, Wk, Wv, Wo, wtq, wtk, wtv, wto);
    // 1) fused Q/K/V projections (grid.z picks which)
    proj_kernel<<<dim3((H * E) / 64, (B * S) / 128, 3), 128, 0, stream>>>(
        q, k, v, wtq, wtk, wtv, qh, kh, vt);
    // 2) flash attention
    attn_kernel<<<dim3(S / 64, B * H), 128, 0, stream>>>(qh, kh, vt, z);
    // 3) output projection + bias
    outproj_kernel<<<dim3(D / 64, (B * S) / 128), 128, 0, stream>>>(z, Wo ? wto : wto, bo, out);
}
